// LocalGNN_t_6777458393749
// MI455X (gfx1250) — matrix operand order repaired and checked by reading, not compiled, not run on hardware
//
#include <hip/hip_runtime.h>
#include <string.h>

// ---------------------------------------------------------------------------
// LocalGNN on MI455X (gfx1250, wave32, WMMA).
// Dominant cost: 4 skinny diffusion GEMMs Z = X @ S with N=K=8192, M in {128,256}.
// Precision: bf16 hi/lo split (3-term WMMA: Xh*Sh + Xh*Sl + Xl*Sh, fp32 acc)
//   -> fp32-class accuracy at bf16 matrix-core rates.
// Occupancy: split-K x8 (grid.y) -> 512 WGs / 4096 waves per diffusion pass;
//   deterministic fixed-order reduction of the 8 partials (no float atomics).
// Data path: S transposed+split once to bf16 (column-major => WMMA B fragment is
//   contiguous per lane, streamed non-temporally exactly once per pass);
//   X chunk staged in LDS per workgroup (ds_load_b128 A fragments).
// Workspace use: ~395 MB (S hi/lo 2x128MB + 64MB split-K partials + misc).
// ---------------------------------------------------------------------------

#define NN 8192
#define KSPLIT 8
#define KLEN (NN / KSPLIT)

typedef __attribute__((ext_vector_type(8)))  __bf16 v8bf;
typedef __attribute__((ext_vector_type(16))) __bf16 v16bf;
typedef __attribute__((ext_vector_type(8)))  float  v8f;

__device__ __forceinline__ __bf16 f2bf(float f) {
    unsigned u = __builtin_bit_cast(unsigned, f);
    unsigned r = u + 0x7FFFu + ((u >> 16) & 1u);   // round-to-nearest-even
    return __builtin_bit_cast(__bf16, (unsigned short)(r >> 16));
}
__device__ __forceinline__ float bf2f(__bf16 b) {
    unsigned short s = __builtin_bit_cast(unsigned short, b);
    return __builtin_bit_cast(float, (unsigned)s << 16);
}
__device__ __forceinline__ v16bf cat16(v8bf a, v8bf b) {
    return __builtin_shufflevector(a, b, 0,1,2,3,4,5,6,7,8,9,10,11,12,13,14,15);
}

// ---------------------------------------------------------------------------
// S (fp32, row-major [k][n]) -> St_hi/St_lo (bf16, transposed [n][k]).
// 64x64 tiles through LDS for coalescing on both sides.
// ---------------------------------------------------------------------------
__global__ void k_transpose_split_S(const float* __restrict__ S,
                                    __bf16* __restrict__ Sth,
                                    __bf16* __restrict__ Stl) {
    __shared__ __bf16 th[64][66];
    __shared__ __bf16 tl[64][66];
    const int k0 = blockIdx.y * 64;
    const int n0 = blockIdx.x * 64;
    for (int i = threadIdx.x; i < 64 * 64; i += 256) {
        int k = i >> 6, n = i & 63;
        float v = S[(size_t)(k0 + k) * NN + (n0 + n)];
        __bf16 h = f2bf(v);
        th[k][n] = h;
        tl[k][n] = f2bf(v - bf2f(h));
    }
    __syncthreads();
    for (int i = threadIdx.x; i < 64 * 64; i += 256) {
        int n = i >> 6, k = i & 63;
        Sth[(size_t)(n0 + n) * NN + (k0 + k)] = th[k][n];
        Stl[(size_t)(n0 + n) * NN + (k0 + k)] = tl[k][n];
    }
}

// ---------------------------------------------------------------------------
// fp32 -> bf16 hi/lo elementwise split (for x).
// ---------------------------------------------------------------------------
__global__ void k_split(const float* __restrict__ X, __bf16* __restrict__ H,
                        __bf16* __restrict__ L, int count) {
    int i = blockIdx.x * blockDim.x + threadIdx.x;
    if (i < count) {
        float v = X[i];
        __bf16 h = f2bf(v);
        H[i] = h;
        L[i] = f2bf(v - bf2f(h));
    }
}

// ---------------------------------------------------------------------------
// Diffusion (split-K partial): Zpart[by][M][NN] = A[M][ks:ks+KLEN] @ S[ks:ks+KLEN][:].
// A given as bf16 hi/lo, S as transposed bf16 hi/lo (St[n][k]).
// 8 waves per WG; wave w owns columns blockIdx.x*128 + 16w .. +16w+15.
// Each wave accumulates all MTILES 16x16 D tiles. A chunk (32 k's) staged in LDS.
// ---------------------------------------------------------------------------
template <int MTILES>
__global__ void __launch_bounds__(256)
k_diffusion(const __bf16* __restrict__ Ah, const __bf16* __restrict__ Al,
            const __bf16* __restrict__ Sth, const __bf16* __restrict__ Stl,
            float* __restrict__ Zpart) {
    constexpr int M    = MTILES * 16;
    constexpr int KC   = 32;   // k per step (one WMMA depth)
    constexpr int KPAD = 40;   // padded LDS row stride (elements): conflict-free
    __shared__ __bf16 lAh[M * KPAD];
    __shared__ __bf16 lAl[M * KPAD];

    const int tid  = threadIdx.x;
    const int wave = tid >> 5;
    const int lane = tid & 31;
    const int half = lane >> 4;   // 0: lanes 0-15, 1: lanes 16-31
    const int l16  = lane & 15;
    const int ks   = blockIdx.y * KLEN;                  // split-K range start
    const int col  = blockIdx.x * 128 + wave * 16 + l16; // output column (B col)

    // B fragment source: St row "col", contiguous in k. Lane half selects K 0-15 / 16-31.
    const __bf16* bh_ptr = Sth + (size_t)col * NN + half * 16;
    const __bf16* bl_ptr = Stl + (size_t)col * NN + half * 16;

    v8f acc[MTILES];
#pragma unroll
    for (int r = 0; r < MTILES; ++r)
        acc[r] = v8f{0.f, 0.f, 0.f, 0.f, 0.f, 0.f, 0.f, 0.f};

    for (int k0 = ks; k0 < ks + KLEN; k0 += KC) {
        // ---- cooperative stage of A chunk (hi & lo) into LDS ----
        __syncthreads();
        constexpr int CHUNKS = M * KC / 8;  // 16-byte chunks per array
        for (int i = tid; i < CHUNKS; i += 256) {
            int row = i >> 2;
            int cg  = i & 3;
            v8bf vh = *(const v8bf*)(Ah + (size_t)row * NN + k0 + cg * 8);
            v8bf vl = *(const v8bf*)(Al + (size_t)row * NN + k0 + cg * 8);
            *(v8bf*)(lAh + row * KPAD + cg * 8) = vh;
            *(v8bf*)(lAl + row * KPAD + cg * 8) = vl;
        }
        __syncthreads();

        // ---- B fragments: 16 contiguous bf16 per lane, streamed non-temporally ----
        v8bf bh0 = __builtin_nontemporal_load((const v8bf*)(bh_ptr + k0));
        v8bf bh1 = __builtin_nontemporal_load((const v8bf*)(bh_ptr + k0 + 8));
        v8bf bl0 = __builtin_nontemporal_load((const v8bf*)(bl_ptr + k0));
        v8bf bl1 = __builtin_nontemporal_load((const v8bf*)(bl_ptr + k0 + 8));
        v16bf b_h = cat16(bh0, bh1);
        v16bf b_l = cat16(bl0, bl1);
        if (k0 + KC < ks + KLEN) {  // prefetch next B chunk (global_prefetch_b8)
            __builtin_prefetch(bh_ptr + k0 + KC, 0, 0);
            __builtin_prefetch(bl_ptr + k0 + KC, 0, 0);
        }

#pragma unroll
        for (int r = 0; r < MTILES; ++r) {
            // A fragment (16x32 bf16): lanes<16 -> K 0-15; lanes>=16 -> K 16-31
            const __bf16* ah = lAh + (r * 16 + l16) * KPAD + half * 16;
            const __bf16* al = lAl + (r * 16 + l16) * KPAD + half * 16;
            v16bf a_h = cat16(*(const v8bf*)(ah), *(const v8bf*)(ah + 8));
            v16bf a_l = cat16(*(const v8bf*)(al), *(const v8bf*)(al + 8));
            acc[r] = __builtin_amdgcn_wmma_f32_16x16x32_bf16(
                false, a_h, false, b_h, (short)0, acc[r], false, false);
            acc[r] = __builtin_amdgcn_wmma_f32_16x16x32_bf16(
                false, a_h, false, b_l, (short)0, acc[r], false, false);
            acc[r] = __builtin_amdgcn_wmma_f32_16x16x32_bf16(
                false, a_l, false, b_h, (short)0, acc[r], false, false);
        }
    }

    // ---- store partial: C/D layout: VGPR j -> row = j + 8*half (within tile), col = l16 ----
    float* Zp = Zpart + (size_t)blockIdx.y * M * NN;
#pragma unroll
    for (int r = 0; r < MTILES; ++r) {
#pragma unroll
        for (int j = 0; j < 8; ++j) {
            int row = r * 16 + half * 8 + j;
            Zp[(size_t)row * NN + col] = acc[r][j];
        }
    }
}

// ---------------------------------------------------------------------------
// Deterministic fixed-order reduction of KSPLIT partials; optionally also
// emits the bf16 hi/lo split of the result for the next diffusion hop.
// ---------------------------------------------------------------------------
__global__ void k_reduce(const float* __restrict__ Zpart, int MN,
                         float* __restrict__ Z, __bf16* __restrict__ Zh,
                         __bf16* __restrict__ Zl) {
    int i = blockIdx.x * blockDim.x + threadIdx.x;
    if (i < MN) {
        float s = 0.f;
#pragma unroll
        for (int p = 0; p < KSPLIT; ++p) s += Zpart[(size_t)p * MN + i];
        Z[i] = s;
        if (Zh != nullptr) {
            __bf16 h = f2bf(s);
            Zh[i] = h;
            Zl[i] = f2bf(s - bf2f(h));
        }
    }
}

// ---------------------------------------------------------------------------
// Layer-1 tap combine: Y[bt,g,n] = tanh( sum_{k,f} h1[g,k,f] z_k[bt,f,n] + b1[g] )
// z rows are (bt*4+f); Y rows are (bt*8+g). Also emits bf16 hi/lo of Y.
// ---------------------------------------------------------------------------
__global__ void k_combine1(const float* __restrict__ z0, const float* __restrict__ z1,
                           const float* __restrict__ z2, const float* __restrict__ h,
                           const float* __restrict__ b, float* __restrict__ Y,
                           __bf16* __restrict__ Yh, __bf16* __restrict__ Yl) {
    __shared__ float sh[96 + 8];  // h1 [8][3][4] + b1 [8]
    for (int i = threadIdx.x; i < 104; i += blockDim.x)
        sh[i] = (i < 96) ? h[i] : b[i - 96];
    __syncthreads();

    int gid = blockIdx.x * blockDim.x + threadIdx.x;  // over 32 * NN
    int bt = gid / NN, n = gid % NN;
    float zv[3][4];
#pragma unroll
    for (int f = 0; f < 4; ++f) {
        size_t o  = (size_t)(bt * 4 + f) * NN + n;
        zv[0][f] = z0[o];
        zv[1][f] = z1[o];
        zv[2][f] = z2[o];
    }
#pragma unroll
    for (int g = 0; g < 8; ++g) {
        float s = sh[96 + g];
#pragma unroll
        for (int k = 0; k < 3; ++k)
#pragma unroll
            for (int f = 0; f < 4; ++f)
                s += sh[(g * 3 + k) * 4 + f] * zv[k][f];
        float y  = tanhf(s);
        size_t o = (size_t)(bt * 8 + g) * NN + n;
        Y[o] = y;
        __bf16 hh = f2bf(y);
        Yh[o] = hh;
        Yl[o] = f2bf(y - bf2f(hh));
    }
}

// ---------------------------------------------------------------------------
// Layer-2 tap combine fused with per-node readout MLP:
// y2[g] = tanh(sum h2[g,k,f] z_k[f] + b2[g]);  u = tanh(W1 y2 + bw1);
// out[bt,e,n] = W2 u + bw2.   Output layout matches [B,T,D1,N] flat.
// ---------------------------------------------------------------------------
__global__ void k_combine2_readout(const float* __restrict__ z0,
                                   const float* __restrict__ z1,
                                   const float* __restrict__ z2,
                                   const float* __restrict__ h2,
                                   const float* __restrict__ b2,
                                   const float* __restrict__ W1,
                                   const float* __restrict__ bw1,
                                   const float* __restrict__ W2,
                                   const float* __restrict__ bw2,
                                   float* __restrict__ out) {
    // layout in LDS: [0..191] h2, [192..199] b2, [200..263] W1, [264..271] bw1,
    //                [272..303] W2, [304..307] bw2
    __shared__ float sc[308];
    for (int i = threadIdx.x; i < 308; i += blockDim.x) {
        float v;
        if (i < 192)        v = h2[i];
        else if (i < 200)   v = b2[i - 192];
        else if (i < 264)   v = W1[i - 200];
        else if (i < 272)   v = bw1[i - 264];
        else if (i < 304)   v = W2[i - 272];
        else                v = bw2[i - 304];
        sc[i] = v;
    }
    __syncthreads();

    int gid = blockIdx.x * blockDim.x + threadIdx.x;  // over 32 * NN
    int bt = gid / NN, n = gid % NN;
    float zv[3][8];
#pragma unroll
    for (int f = 0; f < 8; ++f) {
        size_t o  = (size_t)(bt * 8 + f) * NN + n;
        zv[0][f] = z0[o];
        zv[1][f] = z1[o];
        zv[2][f] = z2[o];
    }
    float y2[8];
#pragma unroll
    for (int g = 0; g < 8; ++g) {
        float s = sc[192 + g];
#pragma unroll
        for (int k = 0; k < 3; ++k)
#pragma unroll
            for (int f = 0; f < 8; ++f)
                s += sc[(g * 3 + k) * 8 + f] * zv[k][f];
        y2[g] = tanhf(s);
    }
    float u[8];
#pragma unroll
    for (int d = 0; d < 8; ++d) {
        float s = sc[264 + d];
#pragma unroll
        for (int f = 0; f < 8; ++f) s += sc[200 + d * 8 + f] * y2[f];
        u[d] = tanhf(s);
    }
#pragma unroll
    for (int e = 0; e < 4; ++e) {
        float s = sc[304 + e];
#pragma unroll
        for (int d = 0; d < 8; ++d) s += sc[272 + e * 8 + d] * u[d];
        out[(size_t)(bt * 4 + e) * NN + n] = s;
    }
}

// ---------------------------------------------------------------------------
// Host-side orchestration.
// ---------------------------------------------------------------------------
extern "C" void kernel_launch(void* const* d_in, const int* in_sizes, int n_in,
                              void* d_out, int out_size, void* d_ws, size_t ws_size,
                              hipStream_t stream) {
    const float* x   = (const float*)d_in[0];   // [2,16,4,8192] = [128][8192]
    const float* S   = (const float*)d_in[1];   // [8192,8192]
    const float* h1  = (const float*)d_in[2];   // [8,3,4]
    const float* b1  = (const float*)d_in[3];   // [8,1]
    const float* h2  = (const float*)d_in[4];   // [8,3,8]
    const float* b2  = (const float*)d_in[5];   // [8,1]
    const float* W1  = (const float*)d_in[6];   // [8,8]
    const float* bw1 = (const float*)d_in[7];   // [8]
    const float* W2  = (const float*)d_in[8];   // [4,8]
    const float* bw2 = (const float*)d_in[9];   // [4]
    float* out = (float*)d_out;                 // [2,16,4,8192]

    // Workspace carve (requires ~395 MB; 256-B aligned slices).
    char* w = (char*)d_ws;
    auto carve = [&](size_t bytes) -> void* {
        void* p = (void*)w;
        w += (bytes + 255) & ~(size_t)255;
        return p;
    };
    const size_t SN = (size_t)NN * NN;    // 64M
    const size_t M1 = (size_t)128 * NN;   // 1M
    const size_t M2 = (size_t)256 * NN;   // 2M
    __bf16* Sth   = (__bf16*)carve(SN * 2);
    __bf16* Stl   = (__bf16*)carve(SN * 2);
    float*  Zpart = (float*)carve((size_t)KSPLIT * M2 * 4);  // reused by all 4 hops
    __bf16* Xh    = (__bf16*)carve(M1 * 2);
    __bf16* Xl    = (__bf16*)carve(M1 * 2);
    float*  Z1    = (float*)carve(M1 * 4);
    __bf16* Z1h   = (__bf16*)carve(M1 * 2);
    __bf16* Z1l   = (__bf16*)carve(M1 * 2);
    float*  Z2    = (float*)carve(M1 * 4);
    float*  Y1    = (float*)carve(M2 * 4);
    __bf16* Y1h   = (__bf16*)carve(M2 * 2);
    __bf16* Y1l   = (__bf16*)carve(M2 * 2);
    float*  Z1b   = (float*)carve(M2 * 4);
    __bf16* Z1bh  = (__bf16*)carve(M2 * 2);
    __bf16* Z1bl  = (__bf16*)carve(M2 * 2);
    float*  Z2b   = (float*)carve(M2 * 4);
    (void)ws_size; (void)in_sizes; (void)n_in; (void)out_size;

    const dim3 dgrid(NN / 128, KSPLIT);  // 64 column strips x 8 K-splits = 512 WGs

    // 1) S -> transposed bf16 hi/lo (done every call; deterministic).
    k_transpose_split_S<<<dim3(NN / 64, NN / 64), 256, 0, stream>>>(S, Sth, Stl);
    // 2) x -> bf16 hi/lo.
    k_split<<<(int)((M1 + 255) / 256), 256, 0, stream>>>(x, Xh, Xl, (int)M1);
    // 3) Layer 1 diffusion hops (split-K partials + deterministic reduce).
    k_diffusion<8><<<dgrid, 256, 0, stream>>>(Xh, Xl, Sth, Stl, Zpart);
    k_reduce<<<(int)(M1 / 256), 256, 0, stream>>>(Zpart, (int)M1, Z1, Z1h, Z1l);
    k_diffusion<8><<<dgrid, 256, 0, stream>>>(Z1h, Z1l, Sth, Stl, Zpart);
    k_reduce<<<(int)(M1 / 256), 256, 0, stream>>>(Zpart, (int)M1, Z2, nullptr, nullptr);
    // 4) Layer 1 combine + tanh -> Y1 (fp32 + bf16 hi/lo).
    k_combine1<<<32 * NN / 256, 256, 0, stream>>>(x, Z1, Z2, h1, b1, Y1, Y1h, Y1l);
    // 5) Layer 2 diffusion hops.
    k_diffusion<16><<<dgrid, 256, 0, stream>>>(Y1h, Y1l, Sth, Stl, Zpart);
    k_reduce<<<(int)(M2 / 256), 256, 0, stream>>>(Zpart, (int)M2, Z1b, Z1bh, Z1bl);
    k_diffusion<16><<<dgrid, 256, 0, stream>>>(Z1bh, Z1bl, Sth, Stl, Zpart);
    k_reduce<<<(int)(M2 / 256), 256, 0, stream>>>(Zpart, (int)M2, Z2b, nullptr, nullptr);
    // 6) Layer 2 combine + readout MLP -> out.
    k_combine2_readout<<<32 * NN / 256, 256, 0, stream>>>(
        Y1, Z1b, Z2b, h2, b2, W1, bw1, W2, bw2, out);
}
